// SampleGAT_46514495816186
// MI455X (gfx1250) — matrix operand order repaired and checked by reading, not compiled, not run on hardware
//
#include <hip/hip_runtime.h>
#include <hip/hip_bf16.h>
#include <math.h>

#define E_SZv   128
#define HEADSv  2
#define HCv     64
#define LAYERSv 6
#define G_NUM   1024
#define NPGv    64
#define N_NODES (G_NUM * NPGv)     // 65536
#define NE_EDGES (8 * N_NODES)     // 524288

typedef _Float16 h16;
typedef _Float16 v16h __attribute__((ext_vector_type(16)));
typedef _Float16 h8v  __attribute__((ext_vector_type(8)));
typedef float    v8f  __attribute__((ext_vector_type(8)));

__device__ __forceinline__ float lrelu(float x, float s) { return x > 0.f ? x : s * x; }
__device__ __forceinline__ int   fOrd(float f) { int i = __float_as_int(f); return i >= 0 ? i : (i ^ 0x7fffffff); }
__device__ __forceinline__ float fUnord(int i) { return __int_as_float(i >= 0 ? i : (i ^ 0x7fffffff)); }

// A-fragment: 16 contiguous halfs per lane (K = k0 + (lane/16)*16 + e)
__device__ __forceinline__ v16h mk_afrag(const h16* base) {
    h8v a0 = *(const h8v*)base;
    h8v a1 = *(const h8v*)(base + 8);
    v16h f;
#pragma unroll
    for (int i = 0; i < 8; i++) { f[i] = a0[i]; f[i + 8] = a1[i]; }
    return f;
}
// B-fragment: 16 contiguous halfs per lane (K = k0 + (lane/16)*16 + e)
__device__ __forceinline__ v16h mk_bfrag(const h16* base) {
    h8v b0 = *(const h8v*)base;
    h8v b1 = *(const h8v*)(base + 8);
    v16h f;
#pragma unroll
    for (int i = 0; i < 8; i++) { f[i] = b0[i]; f[i + 8] = b1[i]; }
    return f;
}

// ---------------- generic WMMA GEMM ----------------
// C[M,N] = act(A[M,K] @ W[K,N] + bias); A,W f16 row-major.
// block = 256 threads (8 waves); block tile = 128 rows x 64 cols; grid = (M/128, N/64)
// act: 0 none, 1 relu, 2 leaky(0.01)
__global__ __launch_bounds__(256) void k_gemm(
    const h16* __restrict__ A, const h16* __restrict__ W, const float* __restrict__ bias,
    float* __restrict__ Cf, h16* __restrict__ Ch, h16* __restrict__ Ct,
    int N, int K, int act)
{
    __shared__ alignas(16) h16 Wt[64 * 32]; // transposed weight chunk [n][k]
    const int tid = threadIdx.x;
    const int lane = tid & 31, wave = tid >> 5;
    const int grp = lane >> 4, l16 = lane & 15;
    const int rowBase = blockIdx.x * 128 + wave * 16;
    const int colBase = blockIdx.y * 64;
    // fill assignment: thread -> (k-row, 8-wide n segment); one coalesced b128 load each
    const int fk = tid >> 3;          // 0..31
    const int fn = (tid & 7) * 8;     // 0..56
    v8f acc[4] = {};
    for (int k0 = 0; k0 < K; k0 += 32) {
        h8v w8 = *(const h8v*)&W[(size_t)(k0 + fk) * N + colBase + fn];
#pragma unroll
        for (int j = 0; j < 8; j++) Wt[(fn + j) * 32 + fk] = w8[j];
        __syncthreads();
        const h16* arow = A + (size_t)(rowBase + l16) * K + k0 + grp * 16;
        __builtin_prefetch(arow + 32, 0, 1);
        v16h af = mk_afrag(arow);
#pragma unroll
        for (int t = 0; t < 4; t++) {
            v16h bf = mk_bfrag(&Wt[(t * 16 + l16) * 32 + grp * 16]);
            acc[t] = __builtin_amdgcn_wmma_f32_16x16x32_f16(false, af, false, bf,
                                                            (short)0, acc[t], false, false);
        }
        __syncthreads();
    }
#pragma unroll
    for (int t = 0; t < 4; t++) {
#pragma unroll
        for (int v = 0; v < 8; v++) {
            int r = rowBase + v + grp * 8;
            int c = colBase + t * 16 + l16;
            float val = acc[t][v];
            if (bias) val += bias[c];
            if (act == 1) val = fmaxf(val, 0.f);
            else if (act == 2) val = lrelu(val, 0.01f);
            size_t idx = (size_t)r * N + c;
            if (Cf) Cf[idx] = val;
            if (Ch) Ch[idx] = (h16)val;
            if (Ct) Ct[((size_t)(r >> 6) * N + c) * 64 + (r & 63)] = (h16)val; // per-graph transpose
        }
    }
}

// C[M,16] = A[M,128] @ Wp[128,16] (f16 in, f32 out, no bias/act)
// block = 256 (8 waves); each wave one 16-row tile; grid = M/128
__global__ __launch_bounds__(256) void k_gemm_n16(
    const h16* __restrict__ A, const h16* __restrict__ Wp, float* __restrict__ Cf)
{
    __shared__ alignas(16) h16 Wt[16 * 128]; // [n][k], full K=128
    const int tid = threadIdx.x;
    const int lane = tid & 31, wave = tid >> 5;
    const int grp = lane >> 4, l16 = lane & 15;
    const int rowBase = blockIdx.x * 128 + wave * 16;
    {
        int k = tid >> 1, n8 = (tid & 1) * 8;
        h8v w8 = *(const h8v*)&Wp[(size_t)k * 16 + n8];
#pragma unroll
        for (int j = 0; j < 8; j++) Wt[(n8 + j) * 128 + k] = w8[j];
    }
    __syncthreads();
    v8f acc = {};
    const h16* arow0 = A + (size_t)(rowBase + l16) * 128;
#pragma unroll
    for (int k0 = 0; k0 < 128; k0 += 32) {
        v16h af = mk_afrag(arow0 + k0 + grp * 16);
        v16h bf = mk_bfrag(&Wt[l16 * 128 + k0 + grp * 16]);
        acc = __builtin_amdgcn_wmma_f32_16x16x32_f16(false, af, false, bf, (short)0, acc, false, false);
    }
#pragma unroll
    for (int v = 0; v < 8; v++)
        Cf[(size_t)(rowBase + v + grp * 8) * 16 + l16] = acc[v];
}

// scores[g][n][m] = sum_c k[g*64+n][c]*q[g*64+m][c]; one wave per 16x16 tile; grid = G*2 blocks
__global__ __launch_bounds__(256) void k_scores(const h16* __restrict__ kx,
                                                const h16* __restrict__ qx,
                                                float* __restrict__ scores)
{
    const int tid = threadIdx.x, lane = tid & 31, wave = tid >> 5;
    const int tileId = blockIdx.x * 8 + wave; // G*16 tiles
    const int g = tileId >> 4;
    const int tn = (tileId >> 2) & 3, tm = tileId & 3;
    const int grp = lane >> 4, l16 = lane & 15;
    v8f acc = {};
    const h16* kbase = kx + ((size_t)g * 64 + tn * 16 + l16) * 128;
    const h16* qbase = qx + ((size_t)g * 64 + tm * 16 + l16) * 128;
#pragma unroll
    for (int k0 = 0; k0 < 128; k0 += 32) {
        v16h af = mk_afrag(kbase + k0 + grp * 16);
        v16h bf = mk_bfrag(qbase + k0 + grp * 16);
        acc = __builtin_amdgcn_wmma_f32_16x16x32_f16(false, af, false, bf, (short)0, acc, false, false);
    }
#pragma unroll
    for (int v = 0; v < 8; v++)
        scores[((size_t)g * 64 + tn * 16 + v + grp * 8) * 64 + tm * 16 + l16] = acc[v];
}

// z[g*64+n][c] = sum_m P[g][n][m] * v[g*64+m][c] via vT[g][c][m]; grid = G*4 blocks
__global__ __launch_bounds__(256) void k_zmat(const h16* __restrict__ P,
                                              const h16* __restrict__ vT,
                                              float* __restrict__ z)
{
    const int tid = threadIdx.x, lane = tid & 31, wave = tid >> 5;
    const int tileId = blockIdx.x * 8 + wave; // G*32 tiles
    const int g = tileId >> 5;
    const int rest = tileId & 31;
    const int tn = rest >> 3, tc = rest & 7;
    const int grp = lane >> 4, l16 = lane & 15;
    v8f acc = {};
    const h16* pbase = P + ((size_t)g * 64 + tn * 16 + l16) * 64;
    const h16* vbase = vT + ((size_t)g * 128 + tc * 16 + l16) * 64;
#pragma unroll
    for (int k0 = 0; k0 < 64; k0 += 32) {
        v16h af = mk_afrag(pbase + k0 + grp * 16);
        v16h bf = mk_bfrag(vbase + k0 + grp * 16);
        acc = __builtin_amdgcn_wmma_f32_16x16x32_f16(false, af, false, bf, (short)0, acc, false, false);
    }
#pragma unroll
    for (int v = 0; v < 8; v++)
        z[((size_t)g * 64 + tn * 16 + v + grp * 8) * 128 + tc * 16 + l16] = acc[v];
}

// ---------------- elementwise / scatter kernels ----------------
__global__ void k_f32_to_f16(const float* s, h16* d, int n) {
    int i = blockIdx.x * 256 + threadIdx.x; if (i < n) d[i] = (h16)s[i];
}
__global__ void k_zero_f32(float* p, int n) {
    int i = blockIdx.x * 256 + threadIdx.x; if (i < n) p[i] = 0.f;
}
__global__ void k_fill_i32(int* p, int v, int n) {
    int i = blockIdx.x * 256 + threadIdx.x; if (i < n) p[i] = v;
}

__global__ void k_node_mlp(const float* __restrict__ x, const float* __restrict__ w,
                           const float* __restrict__ b, float* __restrict__ o32, h16* __restrict__ o16) {
    int i = blockIdx.x * 256 + threadIdx.x;
    if (i >= N_NODES * E_SZv) return;
    int n = i >> 7, c = i & 127;
    float acc = b[c];
#pragma unroll
    for (int k = 0; k < 6; k++) acc += x[n * 6 + k] * w[k * E_SZv + c];
    o32[i] = acc; o16[i] = (h16)acc;
}

__global__ void k_edge_gather(const int* __restrict__ eattr, const float* __restrict__ tab,
                              h16* __restrict__ ea, int rowOff, int rows) {
    int i = blockIdx.x * 256 + threadIdx.x;
    if (i >= rows * E_SZv) return;
    int r = i >> 7, c = i & 127;
    ea[i] = (h16)tab[eattr[rowOff + r] * E_SZv + c];
}

// Wpad[k][j] (f16, [128,16]): j = l*2+h for j<12, zero-padded j>=12
__global__ void k_fold_wae(const float* __restrict__ gwe, const float* __restrict__ gae,
                           h16* __restrict__ wpad) {
    int i = blockIdx.x * 256 + threadIdx.x;
    if (i >= E_SZv * 16) return;
    int j = i & 15, k = i >> 4;
    float a = 0.f;
    if (j < 12) {
        int l = j >> 1, h = j & 1;
        for (int c = 0; c < HCv; c++)
            a += gwe[((size_t)(l * E_SZv + k)) * E_SZv + h * HCv + c] * gae[(l * HEADSv + h) * HCv + c];
    }
    wpad[i] = (h16)a;
}

__global__ void k_loop_scatter(const h16* __restrict__ ea, const int* __restrict__ edges,
                               float* __restrict__ acc, float* __restrict__ cnt) {
    int i = blockIdx.x * 256 + threadIdx.x;
    if (i >= NE_EDGES * E_SZv) return;
    int e = i >> 7, c = i & 127;
    int dst = edges[NE_EDGES + e];
    atomicAdd(&acc[(size_t)dst * E_SZv + c], (float)ea[i]);
    if (c == 0) atomicAdd(&cnt[dst], 1.f);
}
__global__ void k_loop_fin(const float* __restrict__ acc, const float* __restrict__ cnt,
                           h16* __restrict__ out16) {
    int i = blockIdx.x * 256 + threadIdx.x;
    if (i >= N_NODES * E_SZv) return;
    out16[i] = (h16)(acc[i] / fmaxf(cnt[i >> 7], 1.f));
}

__global__ void k_asad(const float* __restrict__ xp, const float* __restrict__ gas,
                       const float* __restrict__ gad, float* __restrict__ a_s,
                       float* __restrict__ a_d, int l) {
    int i = blockIdx.x * 256 + threadIdx.x;
    if (i >= N_NODES * 2) return;
    int n = i >> 1, h = i & 1;
    const float* xr = xp + (size_t)n * E_SZv + h * HCv;
    float s = 0.f, d = 0.f;
    for (int c = 0; c < HCv; c++) {
        s += xr[c] * gas[(l * HEADSv + h) * HCv + c];
        d += xr[c] * gad[(l * HEADSv + h) * HCv + c];
    }
    a_s[i] = s; a_d[i] = d;
}

__global__ void k_edge_max(const int* __restrict__ edges, const float* __restrict__ a_s,
                           const float* __restrict__ a_d, const float* __restrict__ AE,
                           float* __restrict__ alpha, int* __restrict__ m_int, int l) {
    int i = blockIdx.x * 256 + threadIdx.x;
    if (i >= NE_EDGES * 2) return;
    int e = i >> 1, h = i & 1;
    int s = edges[e], d = edges[NE_EDGES + e];
    float a = lrelu(a_s[s * 2 + h] + a_d[d * 2 + h] + AE[(size_t)e * 16 + l * 2 + h], 0.2f);
    alpha[i] = a;
    atomicMax(&m_int[d * 2 + h], fOrd(a));
}

__global__ void k_node_mfin(const float* __restrict__ a_s, const float* __restrict__ a_d,
                            const float* __restrict__ AL, const int* __restrict__ m_int,
                            float* __restrict__ aloop, float* __restrict__ m, int l) {
    int i = blockIdx.x * 256 + threadIdx.x;
    if (i >= N_NODES * 2) return;
    int n = i >> 1, h = i & 1;
    float al = lrelu(a_s[i] + a_d[i] + AL[(size_t)n * 16 + l * 2 + h], 0.2f);
    aloop[i] = al;
    m[i] = fmaxf(fUnord(m_int[i]), al);
}

__global__ void k_edge_pe(const int* __restrict__ edges, const float* __restrict__ alpha,
                          const float* __restrict__ m, float* __restrict__ pe,
                          float* __restrict__ den) {
    int i = blockIdx.x * 256 + threadIdx.x;
    if (i >= NE_EDGES * 2) return;
    int e = i >> 1, h = i & 1;
    int d = edges[NE_EDGES + e];
    float p = expf(alpha[i] - m[d * 2 + h]);
    pe[i] = p;
    atomicAdd(&den[d * 2 + h], p);
}

__global__ void k_edge_scatter(const int* __restrict__ edges, const float* __restrict__ pe,
                               const float* __restrict__ xp, float* __restrict__ msg) {
    int i = blockIdx.x * 256 + threadIdx.x;
    if (i >= NE_EDGES * E_SZv) return;
    int e = i >> 7, c = i & 127, h = c >> 6;
    int s = edges[e], d = edges[NE_EDGES + e];
    atomicAdd(&msg[(size_t)d * E_SZv + c], pe[e * 2 + h] * xp[(size_t)s * E_SZv + c]);
}

__global__ void k_node_combine(const float* __restrict__ msg, const float* __restrict__ den,
                               const float* __restrict__ aloop, const float* __restrict__ m,
                               const float* __restrict__ xp, const float* __restrict__ gat_b,
                               const float* __restrict__ lng, const float* __restrict__ lnb,
                               float* __restrict__ x32, h16* __restrict__ x16, int l, int doLn) {
    int n = blockIdx.x * 256 + threadIdx.x;
    if (n >= N_NODES) return;
    float pl[2], dt[2];
#pragma unroll
    for (int h = 0; h < 2; h++) {
        pl[h] = expf(aloop[n * 2 + h] - m[n * 2 + h]);
        dt[h] = den[n * 2 + h] + pl[h];
    }
    const float* mr = msg + (size_t)n * E_SZv;
    const float* xr = xp + (size_t)n * E_SZv;
    float sum = 0.f, sq = 0.f;
    for (int c = 0; c < E_SZv; c++) {
        int h = c >> 6;
        float v = (mr[c] + pl[h] * xr[c]) / dt[h] + gat_b[l * E_SZv + c];
        sum += v; sq += v * v;
    }
    float* xo = x32 + (size_t)n * E_SZv;
    h16* ho = x16 + (size_t)n * E_SZv;
    if (doLn) {
        float mean = sum * (1.f / E_SZv);
        float var = sq * (1.f / E_SZv) - mean * mean;
        float r = rsqrtf(var + 1e-5f);
        for (int c = 0; c < E_SZv; c++) {
            int h = c >> 6;
            float v = (mr[c] + pl[h] * xr[c]) / dt[h] + gat_b[l * E_SZv + c];
            v = (v - mean) * r * lng[l * E_SZv + c] + lnb[l * E_SZv + c];
            v = fmaxf(v, 0.f);
            xo[c] = v; ho[c] = (h16)v;
        }
    } else {
        for (int c = 0; c < E_SZv; c++) {
            int h = c >> 6;
            float v = (mr[c] + pl[h] * xr[c]) / dt[h] + gat_b[l * E_SZv + c];
            xo[c] = v; ho[c] = (h16)v;
        }
    }
}

__global__ void k_softmax(const float* __restrict__ scores, h16* __restrict__ P) {
    int r = blockIdx.x * 256 + threadIdx.x;
    if (r >= N_NODES) return;
    const float* row = scores + (size_t)r * 64;
    const float sc = 0.08838834764831845f; // 1/sqrt(128)
    float mx = -INFINITY;
    for (int i = 0; i < 64; i++) mx = fmaxf(mx, row[i] * sc);
    float s = 0.f;
    for (int i = 0; i < 64; i++) s += expf(row[i] * sc - mx);
    float inv = 1.f / s;
    h16* pr = P + (size_t)r * 64;
    for (int i = 0; i < 64; i++) pr[i] = (h16)(expf(row[i] * sc - mx) * inv);
}

__global__ void k_attn_ln(const float* __restrict__ x, const float* __restrict__ z,
                          const float* __restrict__ g, const float* __restrict__ b,
                          float* __restrict__ x32, h16* __restrict__ x16) {
    int n = blockIdx.x * 256 + threadIdx.x;
    if (n >= N_NODES) return;
    const float* xr = x + (size_t)n * E_SZv;
    const float* zr = z + (size_t)n * E_SZv;
    float sum = 0.f, sq = 0.f;
    for (int c = 0; c < E_SZv; c++) { float v = xr[c] + zr[c]; sum += v; sq += v * v; }
    float mean = sum * (1.f / E_SZv);
    float var = sq * (1.f / E_SZv) - mean * mean;
    float r = rsqrtf(var + 1e-5f);
    float* xo = x32 + (size_t)n * E_SZv;
    h16* ho = x16 + (size_t)n * E_SZv;
    for (int c = 0; c < E_SZv; c++) {
        float v = ((xr[c] + zr[c]) - mean) * r * g[c] + b[c];
        xo[c] = v; ho[c] = (h16)v;
    }
}

__global__ void k_pool(const h16* __restrict__ h2, h16* __restrict__ gf) {
    int i = blockIdx.x * 256 + threadIdx.x;
    if (i >= G_NUM * 512) return;
    int g = i >> 9, c = i & 511;
    float mx = -INFINITY;
    for (int m2 = 0; m2 < NPGv; m2++)
        mx = fmaxf(mx, (float)h2[((size_t)(g * NPGv + m2)) * 512 + c]);
    gf[i] = (h16)mx;
}

__global__ void k_final(const h16* __restrict__ p2, const float* __restrict__ pw3,
                        const float* __restrict__ pb3, float* __restrict__ out) {
    int g = blockIdx.x * 256 + threadIdx.x;
    if (g >= G_NUM) return;
    float a = pb3[0];
    for (int c = 0; c < E_SZv; c++) a += (float)p2[(size_t)g * E_SZv + c] * pw3[c];
    out[g] = a;
}

// ---------------- host launcher ----------------
extern "C" void kernel_launch(void* const* d_in, const int* in_sizes, int n_in,
                              void* d_out, int out_size, void* d_ws, size_t ws_size,
                              hipStream_t stream) {
    (void)in_sizes; (void)n_in; (void)out_size; (void)ws_size;
    const float* x_in    = (const float*)d_in[0];
    const int*   edges   = (const int*)d_in[1];
    const int*   eattr   = (const int*)d_in[2];
    const float* node_w  = (const float*)d_in[4];
    const float* node_b  = (const float*)d_in[5];
    const float* emb_tab = (const float*)d_in[6];
    const float* ew1     = (const float*)d_in[7];
    const float* eb1     = (const float*)d_in[8];
    const float* ew2     = (const float*)d_in[9];
    const float* eb2     = (const float*)d_in[10];
    const float* gat_w   = (const float*)d_in[11];
    const float* gat_we  = (const float*)d_in[12];
    const float* gat_as  = (const float*)d_in[13];
    const float* gat_ad  = (const float*)d_in[14];
    const float* gat_ae  = (const float*)d_in[15];
    const float* gat_b   = (const float*)d_in[16];
    const float* gat_lng = (const float*)d_in[17];
    const float* gat_lnb = (const float*)d_in[18];
    const float* kw = (const float*)d_in[19]; const float* kb = (const float*)d_in[20];
    const float* qw = (const float*)d_in[21]; const float* qb = (const float*)d_in[22];
    const float* vw = (const float*)d_in[23]; const float* vb = (const float*)d_in[24];
    const float* att_g = (const float*)d_in[25]; const float* att_b = (const float*)d_in[26];
    const float* n2w1 = (const float*)d_in[27]; const float* n2b1 = (const float*)d_in[28];
    const float* n2w2 = (const float*)d_in[29]; const float* n2b2 = (const float*)d_in[30];
    const float* pw1 = (const float*)d_in[31]; const float* pb1 = (const float*)d_in[32];
    const float* pw2 = (const float*)d_in[33]; const float* pb2 = (const float*)d_in[34];
    const float* pw3 = (const float*)d_in[35]; const float* pb3 = (const float*)d_in[36];
    float* out = (float*)d_out;

    char* WS = (char*)d_ws;
    size_t off = 0;
    auto bump = [&](size_t bytes) { size_t a = off; off += (bytes + 255) & ~(size_t)255; return a; };
    auto Fp = [&](size_t o) { return (float*)(WS + o); };
    auto Hp = [&](size_t o) { return (h16*)(WS + o); };
    auto Ip = [&](size_t o) { return (int*)(WS + o); };

    const size_t o_ew1h  = bump(128 * 128 * 2);
    const size_t o_ew2h  = bump(128 * 128 * 2);
    const size_t o_gatwh = bump((size_t)LAYERSv * 128 * 128 * 2);
    const size_t o_kwh   = bump(128 * 128 * 2);
    const size_t o_qwh   = bump(128 * 128 * 2);
    const size_t o_vwh   = bump(128 * 128 * 2);
    const size_t o_n2w1h = bump(128 * 256 * 2);
    const size_t o_n2w2h = bump((size_t)256 * 512 * 2);
    const size_t o_pw1h  = bump((size_t)512 * 256 * 2);
    const size_t o_pw2h  = bump((size_t)256 * 128 * 2);
    const size_t o_wpad  = bump((size_t)E_SZv * 16 * 2);      // folded+padded attn weights f16
    const size_t o_x32   = bump((size_t)N_NODES * 128 * 4);
    const size_t o_x16   = bump((size_t)N_NODES * 128 * 2);
    const size_t o_AE    = bump((size_t)NE_EDGES * 16 * 4);
    const size_t o_AL    = bump((size_t)N_NODES * 16 * 4);
    const size_t o_as    = bump((size_t)N_NODES * 2 * 4);
    const size_t o_ad    = bump((size_t)N_NODES * 2 * 4);
    const size_t o_alpha = bump((size_t)NE_EDGES * 2 * 4);
    const size_t o_pe    = bump((size_t)NE_EDGES * 2 * 4);
    const size_t o_mint  = bump((size_t)N_NODES * 2 * 4);
    const size_t o_m     = bump((size_t)N_NODES * 2 * 4);
    const size_t o_aloop = bump((size_t)N_NODES * 2 * 4);
    const size_t o_den   = bump((size_t)N_NODES * 2 * 4);
    const size_t o_cnt   = bump((size_t)N_NODES * 4);
    const size_t o_xp    = bump((size_t)N_NODES * 128 * 4);   // reused as z
    const size_t o_msg   = bump((size_t)N_NODES * 128 * 4);   // reused as k16 + q16
    const size_t o_lacc  = bump((size_t)N_NODES * 128 * 4);   // loop accum; reused as vT + P
    const size_t o_big   = bump((size_t)NE_EDGES * 128 * 2);  // ea; reused scores / h1 / h2
    const size_t o_tmp0  = bump((size_t)65536 * 128 * 2);     // reused loop_ea16, gf/p1/p2
    const size_t o_tmp1  = bump((size_t)65536 * 128 * 2);

    auto blocks = [](long n) { return (unsigned)((n + 255) / 256); };
    auto conv = [&](const float* s, h16* d, long n) {
        k_f32_to_f16<<<blocks(n), 256, 0, stream>>>(s, d, (int)n);
    };

    // --- weight conversion + folding ---
    conv(ew1, Hp(o_ew1h), 128 * 128);
    conv(ew2, Hp(o_ew2h), 128 * 128);
    conv(gat_w, Hp(o_gatwh), (long)LAYERSv * 128 * 128);
    conv(kw, Hp(o_kwh), 128 * 128);
    conv(qw, Hp(o_qwh), 128 * 128);
    conv(vw, Hp(o_vwh), 128 * 128);
    conv(n2w1, Hp(o_n2w1h), 128 * 256);
    conv(n2w2, Hp(o_n2w2h), 256L * 512);
    conv(pw1, Hp(o_pw1h), 512L * 256);
    conv(pw2, Hp(o_pw2h), 256L * 128);
    k_fold_wae<<<blocks(E_SZv * 16), 256, 0, stream>>>(gat_we, gat_ae, Hp(o_wpad));

    // --- node MLP ---
    k_node_mlp<<<blocks((long)N_NODES * 128), 256, 0, stream>>>(x_in, node_w, node_b, Fp(o_x32), Hp(o_x16));

    // --- edge MLP (chunked GEMMs) -> ea f16 in o_big ---
    k_zero_f32<<<blocks((long)N_NODES * 128), 256, 0, stream>>>(Fp(o_lacc), N_NODES * 128);
    k_zero_f32<<<blocks(N_NODES), 256, 0, stream>>>(Fp(o_cnt), N_NODES);
    const int CH = 65536, NCH = NE_EDGES / CH;
    for (int ch = 0; ch < NCH; ch++) {
        int rowOff = ch * CH;
        k_edge_gather<<<blocks((long)CH * 128), 256, 0, stream>>>(eattr, emb_tab, Hp(o_tmp0), rowOff, CH);
        k_gemm<<<dim3(CH / 128, 2), 256, 0, stream>>>(Hp(o_tmp0), Hp(o_ew1h), eb1,
                                                      nullptr, Hp(o_tmp1), nullptr, 128, 128, 1);
        k_gemm<<<dim3(CH / 128, 2), 256, 0, stream>>>(Hp(o_tmp1), Hp(o_ew2h), eb2,
                                                      nullptr, Hp(o_big) + (size_t)rowOff * 128, nullptr, 128, 128, 0);
    }
    // --- self-loop edge attr (segment mean) + folded attention logits (WMMA, N=16) ---
    k_loop_scatter<<<blocks((long)NE_EDGES * 128), 256, 0, stream>>>(Hp(o_big), edges, Fp(o_lacc), Fp(o_cnt));
    k_loop_fin<<<blocks((long)N_NODES * 128), 256, 0, stream>>>(Fp(o_lacc), Fp(o_cnt), Hp(o_tmp0));
    k_gemm_n16<<<NE_EDGES / 128, 256, 0, stream>>>(Hp(o_big), Hp(o_wpad), Fp(o_AE));
    k_gemm_n16<<<N_NODES / 128, 256, 0, stream>>>(Hp(o_tmp0), Hp(o_wpad), Fp(o_AL));

    // --- GAT layers ---
    const int NEG_INF_ORD = (int)0x807FFFFFu; // fOrd(-inf)
    for (int l = 0; l < LAYERSv; l++) {
        k_gemm<<<dim3(N_NODES / 128, 2), 256, 0, stream>>>(Hp(o_x16), Hp(o_gatwh) + (size_t)l * 128 * 128,
                                                           nullptr, Fp(o_xp), nullptr, nullptr, 128, 128, 0);
        k_asad<<<blocks((long)N_NODES * 2), 256, 0, stream>>>(Fp(o_xp), gat_as, gat_ad, Fp(o_as), Fp(o_ad), l);
        k_fill_i32<<<blocks((long)N_NODES * 2), 256, 0, stream>>>(Ip(o_mint), NEG_INF_ORD, N_NODES * 2);
        k_zero_f32<<<blocks((long)N_NODES * 2), 256, 0, stream>>>(Fp(o_den), N_NODES * 2);
        k_zero_f32<<<blocks((long)N_NODES * 128), 256, 0, stream>>>(Fp(o_msg), N_NODES * 128);
        k_edge_max<<<blocks((long)NE_EDGES * 2), 256, 0, stream>>>(edges, Fp(o_as), Fp(o_ad), Fp(o_AE),
                                                                   Fp(o_alpha), Ip(o_mint), l);
        k_node_mfin<<<blocks((long)N_NODES * 2), 256, 0, stream>>>(Fp(o_as), Fp(o_ad), Fp(o_AL),
                                                                   Ip(o_mint), Fp(o_aloop), Fp(o_m), l);
        k_edge_pe<<<blocks((long)NE_EDGES * 2), 256, 0, stream>>>(edges, Fp(o_alpha), Fp(o_m), Fp(o_pe), Fp(o_den));
        k_edge_scatter<<<blocks((long)NE_EDGES * 128), 256, 0, stream>>>(edges, Fp(o_pe), Fp(o_xp), Fp(o_msg));
        int doLn = (l < LAYERSv - 1) ? 1 : 0;
        const float* lng = gat_lng + (size_t)(doLn ? l : 0) * 128;
        const float* lnb = gat_lnb + (size_t)(doLn ? l : 0) * 128;
        k_node_combine<<<blocks(N_NODES), 256, 0, stream>>>(Fp(o_msg), Fp(o_den), Fp(o_aloop), Fp(o_m),
                                                            Fp(o_xp), gat_b, lng, lnb,
                                                            Fp(o_x32), Hp(o_x16), l, doLn);
    }

    // --- per-graph self-attention (WMMA) ---
    h16* k16 = Hp(o_msg);
    h16* q16 = Hp(o_msg + (size_t)N_NODES * 128 * 2);
    h16* vT  = Hp(o_lacc);
    h16* P   = Hp(o_lacc + (size_t)G_NUM * 128 * 64 * 2);
    float* scores = Fp(o_big);
    float* z = Fp(o_xp);
    k_gemm<<<dim3(N_NODES / 128, 2), 256, 0, stream>>>(Hp(o_x16), Hp(o_kwh), kb, nullptr, k16, nullptr, 128, 128, 0);
    k_gemm<<<dim3(N_NODES / 128, 2), 256, 0, stream>>>(Hp(o_x16), Hp(o_qwh), qb, nullptr, q16, nullptr, 128, 128, 0);
    k_gemm<<<dim3(N_NODES / 128, 2), 256, 0, stream>>>(Hp(o_x16), Hp(o_vwh), vb, nullptr, nullptr, vT, 128, 128, 0);
    k_scores<<<G_NUM * 2, 256, 0, stream>>>(k16, q16, scores);
    k_softmax<<<blocks(N_NODES), 256, 0, stream>>>(scores, P);
    k_zmat<<<G_NUM * 4, 256, 0, stream>>>(P, vT, z);
    k_attn_ln<<<blocks(N_NODES), 256, 0, stream>>>(Fp(o_x32), z, att_g, att_b, Fp(o_x32), Hp(o_x16));

    // --- node2Ef MLP + pool + predictor ---
    h16* h1 = Hp(o_big);
    h16* h2 = Hp(o_big + (size_t)N_NODES * 256 * 2);
    k_gemm<<<dim3(N_NODES / 128, 4), 256, 0, stream>>>(Hp(o_x16), Hp(o_n2w1h), n2b1, nullptr, h1, nullptr, 256, 128, 1);
    k_gemm<<<dim3(N_NODES / 128, 8), 256, 0, stream>>>(h1, Hp(o_n2w2h), n2b2, nullptr, h2, nullptr, 512, 256, 0);
    h16* gf = Hp(o_tmp0);
    h16* p1 = Hp(o_tmp0 + (size_t)G_NUM * 512 * 2);
    h16* p2 = Hp(o_tmp0 + (size_t)G_NUM * 512 * 2 + (size_t)G_NUM * 256 * 2);
    k_pool<<<blocks((long)G_NUM * 512), 256, 0, stream>>>(h2, gf);
    k_gemm<<<dim3(G_NUM / 128, 4), 256, 0, stream>>>(gf, Hp(o_pw1h), pb1, nullptr, p1, nullptr, 256, 512, 2);
    k_gemm<<<dim3(G_NUM / 128, 2), 256, 0, stream>>>(p1, Hp(o_pw2h), pb2, nullptr, p2, nullptr, 128, 256, 2);
    k_final<<<blocks(G_NUM), 256, 0, stream>>>(p2, pw3, pb3, out);
}
